// RelationAwareBias_18863496364256
// MI455X (gfx1250) — compile-verified, matrix operand order checked by reading, not run on hardware
//
#include <hip/hip_runtime.h>
#include <hip/hip_bf16.h>
#include <stdint.h>

#define B_    2
#define H_    12
#define L_    512
#define D_    64
#define R_    8
#define RANK_ 8

typedef __attribute__((ext_vector_type(16))) __bf16         bf16x16;
typedef __attribute__((ext_vector_type(8)))  float          f32x8;
typedef __attribute__((ext_vector_type(8)))  unsigned short u16x8;

union BfTile {
    bf16x16 v;
    u16x8   h[2];
};

__device__ __forceinline__ unsigned short f32_to_bf16(float f) {
    unsigned int u = __float_as_uint(f);
    // round-to-nearest-even
    u += 0x7FFFu + ((u >> 16) & 1u);
    return (unsigned short)(u >> 16);
}

// ---------------------------------------------------------------------------
// Pass 1a: convert Q and K (fp32 -> bf16), row-major [B,H,L,D]
// ---------------------------------------------------------------------------
__global__ void convert_qk(const float* __restrict__ Q, const float* __restrict__ K,
                           unsigned short* __restrict__ Qbf, unsigned short* __restrict__ Kbf,
                           int n) {
    int i = blockIdx.x * blockDim.x + threadIdx.x;
    if (i < n)            Qbf[i]     = f32_to_bf16(Q[i]);
    else if (i < 2 * n)   Kbf[i - n] = f32_to_bf16(K[i - n]);
}

// ---------------------------------------------------------------------------
// Pass 1b: combined weights  W[r,h] = projB[r,h] (DxRANK) @ projA[r,h] (RANKxD)
// Stored TRANSPOSED: WT[(rh)*4096 + e*64 + d] = W[d][e]   (e = d_out, d = d_in)
// so WMMA B-operand lanes read contiguous 16B runs over d_in.
// ---------------------------------------------------------------------------
__global__ void build_weights(const float* __restrict__ projA,   // [R,H,RANK,D]
                              const float* __restrict__ projB,   // [R,H,D,RANK]
                              unsigned short* __restrict__ WT) { // [R,H,D(out),D(in)]
    const int rh = blockIdx.x;                     // 0..R*H-1
    const float* A  = projA + (size_t)rh * RANK_ * D_;
    const float* Bm = projB + (size_t)rh * D_ * RANK_;
    unsigned short* W = WT + (size_t)rh * D_ * D_;
    for (int t = threadIdx.x; t < D_ * D_; t += blockDim.x) {
        const int e = t >> 6;      // d_out
        const int d = t & 63;      // d_in
        float s = 0.f;
#pragma unroll
        for (int x = 0; x < RANK_; ++x)
            s += Bm[d * RANK_ + x] * A[x * D_ + e];
        W[e * D_ + d] = f32_to_bf16(s);
    }
}

// ---------------------------------------------------------------------------
// Pass 2: Xa[b,r,h] (Lx64) = Xbf[b,h] (Lx64) @ W[r,h] (64x64), bf16 out.
// One wave handles a 16-row tile for one (b,r,h): 8 WMMAs.
// grid: x = L/16/8, y = B*R*H.  block = 256 (8 waves).
// ---------------------------------------------------------------------------
__global__ void adapt_kernel(const unsigned short* __restrict__ Xbf, // [B,H,L,D] bf16
                             const unsigned short* __restrict__ WT,  // [R,H,Dout,Din] bf16
                             unsigned short* __restrict__ Xa) {      // [B,R,H,L,D] bf16
    const int wave = threadIdx.x >> 5;
    const int lane = threadIdx.x & 31;
    const int half = lane >> 4;
    const int mrow = lane & 15;
    const int ltile = blockIdx.x * 8 + wave;
    const int y = blockIdx.y;
    const int b = y / (R_ * H_);
    const int r = (y / H_) % R_;
    const int h = y % H_;
    const int l0 = ltile * 16;

    // A operand: two 16x32 tiles of X (K = d_in), per-lane contiguous 16B loads.
    const unsigned short* xrow =
        Xbf + ((size_t)((b * H_ + h) * L_) + l0 + mrow) * D_;
    BfTile a0, a1;
    a0.h[0] = *(const u16x8*)(xrow +  0 + half * 8);
    a0.h[1] = *(const u16x8*)(xrow + 16 + half * 8);
    a1.h[0] = *(const u16x8*)(xrow + 32 + half * 8);
    a1.h[1] = *(const u16x8*)(xrow + 48 + half * 8);

    const unsigned short* wbase = WT + (size_t)(r * H_ + h) * D_ * D_;

    f32x8 acc[4];
#pragma unroll
    for (int nt = 0; nt < 4; ++nt) {
        // B operand: lane holds column n = nt*16 + (lane&15); WT row is contiguous over d_in.
        const unsigned short* wcol = wbase + (size_t)(nt * 16 + mrow) * D_;
        BfTile b0, b1;
        b0.h[0] = *(const u16x8*)(wcol +  0 + half * 8);
        b0.h[1] = *(const u16x8*)(wcol + 16 + half * 8);
        b1.h[0] = *(const u16x8*)(wcol + 32 + half * 8);
        b1.h[1] = *(const u16x8*)(wcol + 48 + half * 8);
        f32x8 c = {};
        c = __builtin_amdgcn_wmma_f32_16x16x32_bf16(false, a0.v, false, b0.v,
                                                    (short)0, c, false, false);
        c = __builtin_amdgcn_wmma_f32_16x16x32_bf16(false, a1.v, false, b1.v,
                                                    (short)0, c, false, false);
        acc[nt] = c;
    }

    unsigned short* dst = Xa + (size_t)(((b * R_ + r) * H_ + h) * L_ + l0) * D_;
#pragma unroll
    for (int nt = 0; nt < 4; ++nt) {
#pragma unroll
        for (int j = 0; j < 8; ++j) {
            const int row = j + half * 8;           // C/D layout: M = j + 8*half
            const int col = nt * 16 + mrow;         // N = lane&15
            dst[(size_t)row * D_ + col] = f32_to_bf16(acc[nt][j]);
        }
    }
}

// ---------------------------------------------------------------------------
// Pass 3: per 16x16 (l,m) tile of one (b,h): 8 relation score tiles via WMMA
// (Qa @ Ka^T, K=64 in two x32 steps), then gather by relation_ids.
// grid: x = (L/16)^2 / 8, y = B*H.  block = 256 (8 waves).
// ---------------------------------------------------------------------------
__global__ void scores_gather(const unsigned short* __restrict__ Qa, // [B,R,H,L,D] bf16
                              const unsigned short* __restrict__ Ka, // [B,R,H,L,D] bf16
                              const int* __restrict__ rel,           // [B,L,L]
                              float* __restrict__ out) {             // [B,H,L,L]
    const int wave = threadIdx.x >> 5;
    const int lane = threadIdx.x & 31;
    const int half = lane >> 4;
    const int mrow = lane & 15;
    const int t  = blockIdx.x * 8 + wave;     // tile id, 0..1023
    const int lt = t >> 5;
    const int mt = t & 31;
    const int b  = blockIdx.y / H_;
    const int h  = blockIdx.y % H_;
    const int l0 = lt * 16;
    const int m0 = mt * 16;

    f32x8 acc[R_];
#pragma unroll
    for (int r = 0; r < R_; ++r) {
        const size_t base = (size_t)(((b * R_ + r) * H_ + h)) * L_;
        const unsigned short* qrow = Qa + (base + l0 + mrow) * D_;  // A operand (M = l)
        const unsigned short* krow = Ka + (base + m0 + mrow) * D_;  // B operand (N = m), K = d contiguous
        BfTile a0, a1, bb0, bb1;
        a0.h[0]  = *(const u16x8*)(qrow +  0 + half * 8);
        a0.h[1]  = *(const u16x8*)(qrow + 16 + half * 8);
        a1.h[0]  = *(const u16x8*)(qrow + 32 + half * 8);
        a1.h[1]  = *(const u16x8*)(qrow + 48 + half * 8);
        bb0.h[0] = *(const u16x8*)(krow +  0 + half * 8);
        bb0.h[1] = *(const u16x8*)(krow + 16 + half * 8);
        bb1.h[0] = *(const u16x8*)(krow + 32 + half * 8);
        bb1.h[1] = *(const u16x8*)(krow + 48 + half * 8);
        f32x8 c = {};
        c = __builtin_amdgcn_wmma_f32_16x16x32_bf16(false, a0.v, false, bb0.v,
                                                    (short)0, c, false, false);
        c = __builtin_amdgcn_wmma_f32_16x16x32_bf16(false, a1.v, false, bb1.v,
                                                    (short)0, c, false, false);
        acc[r] = c;
    }

    // Gather the relation selected per (l,m) element and store.
#pragma unroll
    for (int j = 0; j < 8; ++j) {
        const int row = l0 + j + half * 8;
        const int col = m0 + mrow;
        const int rid = rel[((size_t)b * L_ + row) * L_ + col];
        float v = acc[0][j];
#pragma unroll
        for (int r = 1; r < R_; ++r) v = (rid == r) ? acc[r][j] : v;
        out[(((size_t)b * H_ + h) * L_ + row) * L_ + col] = v;
    }
}

// ---------------------------------------------------------------------------
extern "C" void kernel_launch(void* const* d_in, const int* in_sizes, int n_in,
                              void* d_out, int out_size, void* d_ws, size_t ws_size,
                              hipStream_t stream) {
    const float* Q   = (const float*)d_in[0];
    const float* K   = (const float*)d_in[1];
    const int*   rel = (const int*)  d_in[2];
    const float* qA  = (const float*)d_in[3];
    const float* qB  = (const float*)d_in[4];
    const float* kA  = (const float*)d_in[5];
    const float* kB  = (const float*)d_in[6];
    float* out = (float*)d_out;

    const size_t nQK = (size_t)B_ * H_ * L_ * D_;          // 786432
    const size_t nW  = (size_t)R_ * H_ * D_ * D_;          // 393216
    const size_t nA  = (size_t)B_ * R_ * H_ * L_ * D_;     // 6291456

    char* ws = (char*)d_ws;
    unsigned short* Qbf = (unsigned short*)ws;  ws += nQK * sizeof(unsigned short);
    unsigned short* Kbf = (unsigned short*)ws;  ws += nQK * sizeof(unsigned short);
    unsigned short* WqT = (unsigned short*)ws;  ws += nW  * sizeof(unsigned short);
    unsigned short* WkT = (unsigned short*)ws;  ws += nW  * sizeof(unsigned short);
    unsigned short* Qa  = (unsigned short*)ws;  ws += nA  * sizeof(unsigned short);
    unsigned short* Ka  = (unsigned short*)ws;  ws += nA  * sizeof(unsigned short);
    // total workspace: ~29.9 MB

    // Pass 1a: fp32 -> bf16 for Q and K
    {
        const int total = (int)(2 * nQK);
        convert_qk<<<(total + 255) / 256, 256, 0, stream>>>(Q, K, Qbf, Kbf, (int)nQK);
    }
    // Pass 1b: combined (rank-collapsed) weights, stored transposed
    build_weights<<<R_ * H_, 256, 0, stream>>>(qA, qB, WqT);
    build_weights<<<R_ * H_, 256, 0, stream>>>(kA, kB, WkT);

    // Pass 2: Qa / Ka = X @ W  (WMMA bf16)
    {
        dim3 grid(L_ / 16 / 8, B_ * R_ * H_, 1);   // (4, 192)
        adapt_kernel<<<grid, 256, 0, stream>>>(Qbf, WqT, Qa);
        adapt_kernel<<<grid, 256, 0, stream>>>(Kbf, WkT, Ka);
    }

    // Pass 3: scores per relation in-register + gather
    {
        dim3 grid((L_ / 16) * (L_ / 16) / 8, B_ * H_, 1);  // (128, 24)
        scores_gather<<<grid, 256, 0, stream>>>(Qa, Ka, rel, out);
    }
}
